// FraudGNN_78512002171428
// MI455X (gfx1250) — matrix operand order repaired and checked by reading, not compiled, not run on hardware
//
#include <hip/hip_runtime.h>
#include <hip/hip_bf16.h>

typedef __attribute__((ext_vector_type(16))) _Float16 v16h;
typedef __attribute__((ext_vector_type(8)))  float    v8f;

#define DEV_ATOMIC_ADD(p, v) \
  __hip_atomic_fetch_add((p), (v), __ATOMIC_RELAXED, __HIP_MEMORY_SCOPE_AGENT)

// ---------------------------------------------------------------- small utils
__global__ void fill_ones_k(float* __restrict__ p, int n) {
  int i = blockIdx.x * blockDim.x + threadIdx.x;
  if (i < n) p[i] = 1.0f;
}
__global__ void zero_k(float* __restrict__ p, int n) {
  int i = blockIdx.x * blockDim.x + threadIdx.x;
  if (i < n) p[i] = 0.0f;
}
__global__ void deg_edges_k(const int* __restrict__ dst, float* __restrict__ deg, int E) {
  int e = blockIdx.x * blockDim.x + threadIdx.x;
  if (e < E) DEV_ATOMIC_ADD(&deg[dst[e]], 1.0f);
}
__global__ void to_rsqrt_k(float* __restrict__ p, int n) {
  int i = blockIdx.x * blockDim.x + threadIdx.x;
  if (i < n) { float d = p[i]; p[i] = (d > 0.0f) ? rsqrtf(d) : 0.0f; }
}
__global__ void cvt_f16_k(const float* __restrict__ w, _Float16* __restrict__ o, int n) {
  int i = blockIdx.x * blockDim.x + threadIdx.x;
  if (i < n) o[i] = (_Float16)w[i];
}

// ------------------------------------------------- WMMA GEMM: (M x 128) @ (128 x NT*16)
// One wave computes a 16-row strip across all NT 16-col tiles.
// A: fp32 row-major (converted to f16 per the ISA 16-bit A layout)
// Wh: f16 row-major (K x Nout); B-frag: lane l holds col n=l%16, 16 halves in A's k order.
template <int NTILES>
__global__ __launch_bounds__(256)
void gemm_wmma_k128(const float* __restrict__ A, const _Float16* __restrict__ Wh,
                    float* __restrict__ out, int M) {
  const int Nout = NTILES * 16;
  int wave = blockIdx.x * (blockDim.x >> 5) + (threadIdx.x >> 5);
  int lane = threadIdx.x & 31;
  int row0 = wave * 16;
  if (row0 >= M) return;

  int mrow = lane & 15;
  int hi   = lane >> 4;  // 0: lanes 0-15 (K 0-7,16-23), 1: lanes 16-31 (K 8-15,24-31)

  v8f acc[NTILES];
#pragma unroll
  for (int t = 0; t < NTILES; ++t)
#pragma unroll
    for (int i = 0; i < 8; ++i) acc[t][i] = 0.0f;

  const float* arow = A + (size_t)(row0 + mrow) * 128;

#pragma unroll
  for (int k0 = 0; k0 < 128; k0 += 32) {
    const float4* ap0 = (const float4*)(arow + k0 + hi * 8);
    float4 a0 = ap0[0];
    float4 a1 = ap0[1];
    const float4* ap1 = (const float4*)(arow + k0 + hi * 8 + 16);
    float4 a2 = ap1[0];
    float4 a3 = ap1[1];
    v16h afrag;
    afrag[0]  = (_Float16)a0.x; afrag[1]  = (_Float16)a0.y;
    afrag[2]  = (_Float16)a0.z; afrag[3]  = (_Float16)a0.w;
    afrag[4]  = (_Float16)a1.x; afrag[5]  = (_Float16)a1.y;
    afrag[6]  = (_Float16)a1.z; afrag[7]  = (_Float16)a1.w;
    afrag[8]  = (_Float16)a2.x; afrag[9]  = (_Float16)a2.y;
    afrag[10] = (_Float16)a2.z; afrag[11] = (_Float16)a2.w;
    afrag[12] = (_Float16)a3.x; afrag[13] = (_Float16)a3.y;
    afrag[14] = (_Float16)a3.z; afrag[15] = (_Float16)a3.w;

    const _Float16* wrow = Wh + (size_t)(k0 + hi * 8) * Nout + (lane & 15);
#pragma unroll
    for (int t = 0; t < NTILES; ++t) {
      v16h bfrag;
#pragma unroll
      for (int i = 0; i < 8; ++i) {
        bfrag[i]     = wrow[(size_t)i * Nout + t * 16];
        bfrag[8 + i] = wrow[(size_t)(16 + i) * Nout + t * 16];
      }
      acc[t] = __builtin_amdgcn_wmma_f32_16x16x32_f16(
          /*neg_a=*/false, afrag, /*neg_b=*/false, bfrag,
          /*c_mod=*/(short)0, acc[t], /*reuse_a=*/false, /*reuse_b=*/false);
    }
  }

  int ncol  = lane & 15;
  int mbase = row0 + hi * 8;  // lanes 0-15: rows 0-7; lanes 16-31: rows 8-15
#pragma unroll
  for (int t = 0; t < NTILES; ++t)
#pragma unroll
    for (int r = 0; r < 8; ++r)
      out[(size_t)(mbase + r) * Nout + t * 16 + ncol] = acc[t][r];
}

// -------------------------------------------- aggregation: agg = dinv^2 * h (self-loop)
template <int F>
__global__ void self_init_k(const float* __restrict__ h, const float* __restrict__ dinv,
                            float* __restrict__ agg, int total) {
  int i = blockIdx.x * blockDim.x + threadIdx.x;
  if (i >= total) return;
  float di = dinv[i / F];
  agg[i] = h[i] * di * di;
}

// edges: agg[dst] += h[src] * dinv[src]*dinv[dst];  F/4 lanes per edge, float4 gathers
template <int F>
__global__ void edge_agg_k(const int* __restrict__ src, const int* __restrict__ dst,
                           const float* __restrict__ dinv, const float* __restrict__ h,
                           float* __restrict__ agg, int E) {
  const int LPE = F / 4;
  int tid = blockIdx.x * blockDim.x + threadIdx.x;
  int e = tid / LPE;
  int sub = tid - e * LPE;
  if (e >= E) return;
  int s = src[e], d = dst[e];
  float nrm = dinv[s] * dinv[d];
  float4 v = *((const float4*)(h + (size_t)s * F) + sub);
  float* ad = agg + (size_t)d * F + sub * 4;
  DEV_ATOMIC_ADD(ad + 0, v.x * nrm);
  DEV_ATOMIC_ADD(ad + 1, v.y * nrm);
  DEV_ATOMIC_ADD(ad + 2, v.z * nrm);
  DEV_ATOMIC_ADD(ad + 3, v.w * nrm);
}

__global__ void edge_agg2_k(const int* __restrict__ src, const int* __restrict__ dst,
                            const float* __restrict__ dinv, const float* __restrict__ h,
                            float* __restrict__ agg, int E) {
  int e = blockIdx.x * blockDim.x + threadIdx.x;
  if (e >= E) return;
  int s = src[e], d = dst[e];
  float nrm = dinv[s] * dinv[d];
  DEV_ATOMIC_ADD(&agg[(size_t)d * 2 + 0], h[(size_t)s * 2 + 0] * nrm);
  DEV_ATOMIC_ADD(&agg[(size_t)d * 2 + 1], h[(size_t)s * 2 + 1] * nrm);
}

// --------------------------------------------------------------- batch norm (+ReLU)
// One thread per column, blocks stride over row chunks; coalesced reads.
__global__ void bn_partial_k(const float* __restrict__ x, float* __restrict__ sumv,
                             float* __restrict__ sqv, int Nn, int F, int rowsPerBlock) {
  int col = threadIdx.x;  // blockDim.x == F
  int r0 = blockIdx.x * rowsPerBlock;
  int r1 = r0 + rowsPerBlock;
  if (r1 > Nn) r1 = Nn;
  float s = 0.0f, q = 0.0f;
  for (int r = r0; r < r1; ++r) {
    float v = x[(size_t)r * F + col];
    s += v;
    q += v * v;
  }
  DEV_ATOMIC_ADD(&sumv[col], s);
  DEV_ATOMIC_ADD(&sqv[col], q);
}

__global__ void bn_apply_relu_k(const float* __restrict__ x, const float* __restrict__ sumv,
                                const float* __restrict__ sqv, const float* __restrict__ g,
                                const float* __restrict__ beta, float* __restrict__ y,
                                int Nn, int total, int Fmask) {
  int i = blockIdx.x * blockDim.x + threadIdx.x;
  if (i >= total) return;
  int col = i & Fmask;
  float invN = 1.0f / (float)Nn;
  float m = sumv[col] * invN;
  float var = sqv[col] * invN - m * m;
  float t = (x[i] - m) * rsqrtf(var + 1e-5f) * g[col] + beta[col];
  y[i] = t > 0.0f ? t : 0.0f;
}

// -------------------------------------------------- layer 3 GEMM (64 -> 2) + softmax
__global__ void gemm_small_k(const float* __restrict__ A, const float* __restrict__ W3,
                             float* __restrict__ out, int Nn) {
  int i = blockIdx.x * blockDim.x + threadIdx.x;
  if (i >= Nn) return;
  const float* a = A + (size_t)i * 64;
  float s0 = 0.0f, s1 = 0.0f;
#pragma unroll
  for (int k = 0; k < 64; ++k) {
    float v = a[k];
    s0 += v * W3[k * 2 + 0];
    s1 += v * W3[k * 2 + 1];
  }
  out[(size_t)i * 2 + 0] = s0;
  out[(size_t)i * 2 + 1] = s1;
}

__global__ void final_lsm_k(const float* __restrict__ agg, const float* __restrict__ b3,
                            float* __restrict__ out, int Nn) {
  int i = blockIdx.x * blockDim.x + threadIdx.x;
  if (i >= Nn) return;
  float a = agg[(size_t)i * 2 + 0] + b3[0];
  float b = agg[(size_t)i * 2 + 1] + b3[1];
  float m = fmaxf(a, b);
  float lse = m + logf(expf(a - m) + expf(b - m));
  out[(size_t)i * 2 + 0] = a - lse;
  out[(size_t)i * 2 + 1] = b - lse;
}

// ------------------------------------------------------------------------- launch
extern "C" void kernel_launch(void* const* d_in, const int* in_sizes, int n_in,
                              void* d_out, int out_size, void* d_ws, size_t ws_size,
                              hipStream_t stream) {
  const float* x   = (const float*)d_in[0];
  const int*   ei  = (const int*)d_in[1];
  const float* W1  = (const float*)d_in[2];
  // d_in[3] = b1 (cancels in BN), d_in[7] = b2 (cancels in BN)
  const float* g1  = (const float*)d_in[4];
  const float* be1 = (const float*)d_in[5];
  const float* W2  = (const float*)d_in[6];
  const float* g2  = (const float*)d_in[8];
  const float* be2 = (const float*)d_in[9];
  const float* W3  = (const float*)d_in[10];
  const float* b3  = (const float*)d_in[11];
  float* out = (float*)d_out;

  const int N = in_sizes[0] / 128;
  const int E = in_sizes[1] / 2;
  const int* src = ei;
  const int* dst = ei + E;

  // workspace carve-up (256B aligned)
  char* w = (char*)d_ws;
  auto take = [&](size_t bytes) -> void* {
    void* p = (void*)w;
    w += (bytes + 255) & ~(size_t)255;
    return p;
  };
  float*    dinv  = (float*)take((size_t)N * 4);
  float*    bufA  = (float*)take((size_t)N * 128 * 4);
  float*    bufB  = (float*)take((size_t)N * 128 * 4);
  float*    stats = (float*)take(256 * 4);  // sumv[128] | sqv[128]
  float*    sumv  = stats;
  float*    sqv   = stats + 128;
  _Float16* W1h   = (_Float16*)take((size_t)128 * 128 * 2);
  _Float16* W2h   = (_Float16*)take((size_t)128 * 64 * 2);

  const int TPB = 256;
  int nbN  = (N + TPB - 1) / TPB;
  int nbE  = (E + TPB - 1) / TPB;
  int t128 = N * 128, t64 = N * 64, t2 = N * 2;
  int mwaves  = (N + 15) / 16;
  int gblocks = (mwaves + 7) / 8;  // 8 waves / 256-thread block

  // degrees -> dinv = 1/sqrt(deg+1 self loop)
  fill_ones_k<<<nbN, TPB, 0, stream>>>(dinv, N);
  deg_edges_k<<<nbE, TPB, 0, stream>>>(dst, dinv, E);
  to_rsqrt_k<<<nbN, TPB, 0, stream>>>(dinv, N);

  // pre-convert weights to f16 for WMMA B-fragments
  cvt_f16_k<<<(128 * 128 + TPB - 1) / TPB, TPB, 0, stream>>>(W1, W1h, 128 * 128);
  cvt_f16_k<<<(128 * 64 + TPB - 1) / TPB, TPB, 0, stream>>>(W2, W2h, 128 * 64);

  // ---- layer 1: h = x@W1 ; agg ; BN ; relu
  gemm_wmma_k128<8><<<gblocks, TPB, 0, stream>>>(x, W1h, bufA, N);
  self_init_k<128><<<(t128 + TPB - 1) / TPB, TPB, 0, stream>>>(bufA, dinv, bufB, t128);
  edge_agg_k<128><<<(E * 32 + TPB - 1) / TPB, TPB, 0, stream>>>(src, dst, dinv, bufA, bufB, E);
  zero_k<<<1, 256, 0, stream>>>(stats, 256);
  bn_partial_k<<<(N + 255) / 256, 128, 0, stream>>>(bufB, sumv, sqv, N, 128, 256);
  bn_apply_relu_k<<<(t128 + TPB - 1) / TPB, TPB, 0, stream>>>(bufB, sumv, sqv, g1, be1,
                                                              bufA, N, t128, 127);

  // ---- layer 2: h = act1@W2 ; agg ; BN ; relu
  gemm_wmma_k128<4><<<gblocks, TPB, 0, stream>>>(bufA, W2h, bufB, N);
  self_init_k<64><<<(t64 + TPB - 1) / TPB, TPB, 0, stream>>>(bufB, dinv, bufA, t64);
  edge_agg_k<64><<<(E * 16 + TPB - 1) / TPB, TPB, 0, stream>>>(src, dst, dinv, bufB, bufA, E);
  zero_k<<<1, 256, 0, stream>>>(stats, 256);
  bn_partial_k<<<(N + 255) / 256, 64, 0, stream>>>(bufA, sumv, sqv, N, 64, 256);
  bn_apply_relu_k<<<(t64 + TPB - 1) / TPB, TPB, 0, stream>>>(bufA, sumv, sqv, g2, be2,
                                                             bufB, N, t64, 63);

  // ---- layer 3: h = act2@W3 ; agg ; +b3 ; log_softmax
  gemm_small_k<<<nbN, TPB, 0, stream>>>(bufB, W3, bufA, N);
  self_init_k<2><<<(t2 + TPB - 1) / TPB, TPB, 0, stream>>>(bufA, dinv, bufB, t2);
  edge_agg2_k<<<nbE, TPB, 0, stream>>>(src, dst, dinv, bufA, bufB, E);
  final_lsm_k<<<nbN, TPB, 0, stream>>>(bufB, b3, out, N);
}
